// SLA_GCN_76630806496038
// MI455X (gfx1250) — matrix operand order repaired and checked by reading, not compiled, not run on hardware
//
#include <hip/hip_runtime.h>
#include <hip/hip_bf16.h>
#include <stdint.h>

// ---------------------------------------------------------------------------
// GCN forward for MI455X (gfx1250, wave32, WMMA).
//   h1 = relu(Ahat @ (x @ W1) + b1)         Ahat = D^-1/2 (A+I) D^-1/2
//   h2 = relu(Ahat @ (h1 @ W2) + b2)
//   out[g] = mean-pool(h2)[g] @ lin_w + lin_b
// GEMMs use v_wmma_f32_16x16x32_bf16 with a 2-term bf16 split (3 products)
// for ~fp32 accuracy. GEMM epilogue dual-stores p = dis*(X@W) into both the
// gather buffer and the scatter accumulator (self-loop seed), so no copy
// pass is needed. Scatter-add uses f32 global atomics (L2-resident: the
// 51 MB feature matrix fits in the 192 MB L2).
// ---------------------------------------------------------------------------

typedef __attribute__((ext_vector_type(16))) __bf16 v16bf;
typedef __attribute__((ext_vector_type(8)))  float  v8f;

__device__ __forceinline__ uint16_t bf16t(float f) {
    return (uint16_t)(__float_as_uint(f) >> 16);           // truncate to bf16
}
__device__ __forceinline__ float bf16f(uint16_t h) {
    return __uint_as_float(((uint32_t)h) << 16);
}
__device__ __forceinline__ void split2(float f, uint16_t& hi, uint16_t& lo) {
    hi = bf16t(f);
    lo = bf16t(f - bf16f(hi));
}

union AFrag { int4 q[2]; v16bf v; };

// ---- init: deg=1 (self loop), pool=0, cnt=0 --------------------------------
__global__ void k_init(int* __restrict__ deg, float* __restrict__ pool,
                       float* __restrict__ cnt, int N) {
    int i = blockIdx.x * 256 + threadIdx.x;
    if (i < N)       deg[i]  = 1;
    if (i < 64 * 64) pool[i] = 0.f;
    if (i < 64)      cnt[i]  = 0.f;
}

// ---- degree count over edges ----------------------------------------------
__global__ void k_deg(const int* __restrict__ dst, int* __restrict__ deg, int E) {
    int i = blockIdx.x * 256 + threadIdx.x;
    if (i < E) atomicAdd(&deg[dst[i]], 1);
}

// ---- dis = rsqrt(deg) ------------------------------------------------------
__global__ void k_dis(const int* __restrict__ deg, float* __restrict__ dis, int N) {
    int i = blockIdx.x * 256 + threadIdx.x;
    if (i < N) dis[i] = rsqrtf((float)deg[i]);
}

// ---- f32 -> bf16 hi/lo split ----------------------------------------------
__global__ void k_split(const float* __restrict__ x, uint16_t* __restrict__ hi,
                        uint16_t* __restrict__ lo, int n) {
    int i = blockIdx.x * 256 + threadIdx.x;
    if (i >= n) return;
    uint16_t h, l; split2(x[i], h, l);
    hi[i] = h; lo[i] = l;
}

// ---- pack W1/W2 into per-lane WMMA B-fragment order (hi/lo bf16) ----------
// B 32x16 bf16 layout: lanes 0-15 col N=lane, VGPR j<4 = {K=2j, K=2j+1}, j>=4 = {K=8+2j, K=9+2j};
// lanes 16-31 same cols, K += 8.  Flat index: ((tile*4+kc)*32+lane)*8+j.
__global__ void k_pack(const float* __restrict__ W1, const float* __restrict__ W2,
                       uint32_t* __restrict__ p1h, uint32_t* __restrict__ p1l,
                       uint32_t* __restrict__ p2h, uint32_t* __restrict__ p2l) {
    int i = blockIdx.x * 256 + threadIdx.x;
    if (i < 8192) {                       // W1: 128x128, 8 col tiles, 4 K-chunks
        int j = i & 7, lane = (i >> 3) & 31, kc = (i >> 8) & 3, t = i >> 10;
        int c  = t * 16 + (lane & 15);
        int k  = kc * 32 + ((lane & 16) ? 8 : 0) + 2 * j + ((j >= 4) ? 8 : 0);
        float w0 = W1[k * 128 + c], w1v = W1[(k + 1) * 128 + c];
        uint16_t h0, l0, h1, l1; split2(w0, h0, l0); split2(w1v, h1, l1);
        p1h[i] = ((uint32_t)h1 << 16) | h0;
        p1l[i] = ((uint32_t)l1 << 16) | l0;
    } else if (i < 12288) {               // W2: 128x64, 4 col tiles
        int ii = i - 8192;
        int j = ii & 7, lane = (ii >> 3) & 31, kc = (ii >> 8) & 3, t = ii >> 10;
        int c  = t * 16 + (lane & 15);
        int k  = kc * 32 + ((lane & 16) ? 8 : 0) + 2 * j + ((j >= 4) ? 8 : 0);
        float w0 = W2[k * 64 + c], w1v = W2[(k + 1) * 64 + c];
        uint16_t h0, l0, h1, l1; split2(w0, h0, l0); split2(w1v, h1, l1);
        p2h[ii] = ((uint32_t)h1 << 16) | h0;
        p2l[ii] = ((uint32_t)l1 << 16) | l0;
    }
}

// ---- GEMM1: p1[N,128] = dis * (Xs[N,128] @ W1[128,128]); agg1 := p1 -------
// 1 block = 16 rows, 8 waves = 8 col tiles of 16. K=128 in 4 chunks of 32.
__global__ __launch_bounds__(256)
void k_gemm1(const uint16_t* __restrict__ xhi, const uint16_t* __restrict__ xlo,
             const uint32_t* __restrict__ bh, const uint32_t* __restrict__ bl,
             const float* __restrict__ dis, float* __restrict__ out,
             float* __restrict__ agg, int N) {
    int lane = threadIdx.x & 31, t = threadIdx.x >> 5;
    int rowbase = blockIdx.x * 16;
    int m    = rowbase + (lane & 15);             // A row for this lane
    int koff = (lane & 16) ? 8 : 0;               // A-fragment K sublane offset
    v8f acc = {0.f, 0.f, 0.f, 0.f, 0.f, 0.f, 0.f, 0.f};
#pragma unroll
    for (int kc = 0; kc < 4; ++kc) {
        AFrag ah, al, bhf, blf;
        const uint16_t* pah = xhi + (size_t)m * 128 + kc * 32 + koff;
        const uint16_t* pal = xlo + (size_t)m * 128 + kc * 32 + koff;
        ah.q[0] = *(const int4*)(pah);      ah.q[1] = *(const int4*)(pah + 16);
        al.q[0] = *(const int4*)(pal);      al.q[1] = *(const int4*)(pal + 16);
        const uint32_t* pbh = bh + (size_t)((t * 4 + kc) * 32 + lane) * 8;
        const uint32_t* pbl = bl + (size_t)((t * 4 + kc) * 32 + lane) * 8;
        bhf.q[0] = *(const int4*)(pbh);     bhf.q[1] = *(const int4*)(pbh + 4);
        blf.q[0] = *(const int4*)(pbl);     blf.q[1] = *(const int4*)(pbl + 4);
        acc = __builtin_amdgcn_wmma_f32_16x16x32_bf16(false, ah.v, false, bhf.v,
                                                      (short)0, acc, false, false);
        acc = __builtin_amdgcn_wmma_f32_16x16x32_bf16(false, ah.v, false, blf.v,
                                                      (short)0, acc, false, false);
        acc = __builtin_amdgcn_wmma_f32_16x16x32_bf16(false, al.v, false, bhf.v,
                                                      (short)0, acc, false, false);
    }
    int col = t * 16 + (lane & 15);
    int mb  = rowbase + ((lane & 16) ? 8 : 0);    // C layout: lanes 16-31 -> M+8
#pragma unroll
    for (int r = 0; r < 8; ++r) {
        int row = mb + r;
        float v = acc[r] * dis[row];
        out[(size_t)row * 128 + col] = v;         // gather source p
        agg[(size_t)row * 128 + col] = v;         // scatter seed (self loop)
    }
}

// ---- GEMM2: p2[N,64] = dis * (Hs[N,128] @ W2[128,64]); agg2 := p2 ---------
// 1 block = 32 rows: waves 0-3 -> rows [0,16) tiles 0-3, waves 4-7 -> rows [16,32).
__global__ __launch_bounds__(256)
void k_gemm2(const uint16_t* __restrict__ hhi, const uint16_t* __restrict__ hlo,
             const uint32_t* __restrict__ bh, const uint32_t* __restrict__ bl,
             const float* __restrict__ dis, float* __restrict__ out,
             float* __restrict__ agg, int N) {
    int lane = threadIdx.x & 31, wave = threadIdx.x >> 5;
    int t = wave & 3;
    int rowbase = blockIdx.x * 32 + (wave >> 2) * 16;
    int m    = rowbase + (lane & 15);
    int koff = (lane & 16) ? 8 : 0;
    v8f acc = {0.f, 0.f, 0.f, 0.f, 0.f, 0.f, 0.f, 0.f};
#pragma unroll
    for (int kc = 0; kc < 4; ++kc) {
        AFrag ah, al, bhf, blf;
        const uint16_t* pah = hhi + (size_t)m * 128 + kc * 32 + koff;
        const uint16_t* pal = hlo + (size_t)m * 128 + kc * 32 + koff;
        ah.q[0] = *(const int4*)(pah);      ah.q[1] = *(const int4*)(pah + 16);
        al.q[0] = *(const int4*)(pal);      al.q[1] = *(const int4*)(pal + 16);
        const uint32_t* pbh = bh + (size_t)((t * 4 + kc) * 32 + lane) * 8;
        const uint32_t* pbl = bl + (size_t)((t * 4 + kc) * 32 + lane) * 8;
        bhf.q[0] = *(const int4*)(pbh);     bhf.q[1] = *(const int4*)(pbh + 4);
        blf.q[0] = *(const int4*)(pbl);     blf.q[1] = *(const int4*)(pbl + 4);
        acc = __builtin_amdgcn_wmma_f32_16x16x32_bf16(false, ah.v, false, bhf.v,
                                                      (short)0, acc, false, false);
        acc = __builtin_amdgcn_wmma_f32_16x16x32_bf16(false, ah.v, false, blf.v,
                                                      (short)0, acc, false, false);
        acc = __builtin_amdgcn_wmma_f32_16x16x32_bf16(false, al.v, false, bhf.v,
                                                      (short)0, acc, false, false);
    }
    int col = t * 16 + (lane & 15);
    int mb  = rowbase + ((lane & 16) ? 8 : 0);
#pragma unroll
    for (int r = 0; r < 8; ++r) {
        int row = mb + r;
        float v = acc[r] * dis[row];
        out[(size_t)row * 64 + col] = v;
        agg[(size_t)row * 64 + col] = v;
    }
}

// ---- edge scatter: agg[dst] += p[src], 128 features (32 lanes/edge) -------
__global__ void k_scatter128(const int* __restrict__ src, const int* __restrict__ dst,
                             const float* __restrict__ p, float* __restrict__ agg, int E) {
    int gid = blockIdx.x * 256 + threadIdx.x;
    int e = gid >> 5, q = gid & 31;
    if (e >= E) return;
    int s = src[e], d = dst[e];
    float4 v = *(const float4*)(p + (size_t)s * 128 + q * 4);
    float* a = agg + (size_t)d * 128 + q * 4;
    atomicAdd(a + 0, v.x); atomicAdd(a + 1, v.y);
    atomicAdd(a + 2, v.z); atomicAdd(a + 3, v.w);
}

// ---- edge scatter: 64 features (16 lanes/edge) ----------------------------
__global__ void k_scatter64(const int* __restrict__ src, const int* __restrict__ dst,
                            const float* __restrict__ p, float* __restrict__ agg, int E) {
    int gid = blockIdx.x * 256 + threadIdx.x;
    int e = gid >> 4, q = gid & 15;
    if (e >= E) return;
    int s = src[e], d = dst[e];
    float4 v = *(const float4*)(p + (size_t)s * 64 + q * 4);
    float* a = agg + (size_t)d * 64 + q * 4;
    atomicAdd(a + 0, v.x); atomicAdd(a + 1, v.y);
    atomicAdd(a + 2, v.z); atomicAdd(a + 3, v.w);
}

// ---- finalize conv1: relu(dis*agg + b1) -> bf16 split for layer 2 ---------
__global__ void k_fin1(const float* __restrict__ agg, const float* __restrict__ dis,
                       const float* __restrict__ b1, uint16_t* __restrict__ hhi,
                       uint16_t* __restrict__ hlo, int N) {
    int i = blockIdx.x * 256 + threadIdx.x;
    if (i >= N * 128) return;
    int node = i >> 7, f = i & 127;
    float v = fmaxf(dis[node] * agg[i] + b1[f], 0.f);
    uint16_t h, l; split2(v, h, l);
    hhi[i] = h; hlo[i] = l;
}

// ---- finalize conv2 + fused mean-pool accumulation ------------------------
__global__ void k_fin2(const float* __restrict__ agg, const float* __restrict__ dis,
                       const float* __restrict__ b2, const int* __restrict__ batch,
                       float* __restrict__ pool, int N) {
    int i = blockIdx.x * 256 + threadIdx.x;
    if (i >= N * 64) return;
    int node = i >> 6, f = i & 63;
    float v = fmaxf(dis[node] * agg[i] + b2[f], 0.f);
    atomicAdd(&pool[batch[node] * 64 + f], v);
}

// ---- per-graph node counts -------------------------------------------------
__global__ void k_cnt(const int* __restrict__ batch, float* __restrict__ cnt, int N) {
    int i = blockIdx.x * 256 + threadIdx.x;
    if (i < N) atomicAdd(&cnt[batch[i]], 1.f);
}

// ---- head: out[g] = (pool[g] . lin_w) / cnt[g] + lin_b --------------------
__global__ void k_final(const float* __restrict__ pool, const float* __restrict__ cnt,
                        const float* __restrict__ lw, const float* __restrict__ lb,
                        float* __restrict__ out, int G) {
    int g = threadIdx.x;
    if (g >= G) return;
    float s = 0.f;
#pragma unroll 8
    for (int f = 0; f < 64; ++f) s += pool[g * 64 + f] * lw[f];
    out[g] = s / fmaxf(cnt[g], 1.f) + lb[0];
}

extern "C" void kernel_launch(void* const* d_in, const int* in_sizes, int n_in,
                              void* d_out, int out_size, void* d_ws, size_t ws_size,
                              hipStream_t stream) {
    const float* x     = (const float*)d_in[0];
    const int*   ei    = (const int*)  d_in[1];
    const int*   batch = (const int*)  d_in[2];
    const float* W1    = (const float*)d_in[3];
    const float* b1    = (const float*)d_in[4];
    const float* W2    = (const float*)d_in[5];
    const float* b2    = (const float*)d_in[6];
    const float* lw    = (const float*)d_in[7];
    const float* lb    = (const float*)d_in[8];
    float* out = (float*)d_out;

    const int N  = in_sizes[2];          // 100000 (multiple of 32)
    const int E  = in_sizes[1] / 2;      // 1600000
    const int G  = out_size;             // 64
    const int NF = N * 128, NF2 = N * 64;
    const int* esrc = ei;
    const int* edst = ei + E;

    // workspace carve (256B aligned)
    char* w = (char*)d_ws;
    auto carve = [&](size_t bytes) -> char* {
        char* p = w; w += (bytes + 255) & ~(size_t)255; return p;
    };
    float*    dis  = (float*)   carve((size_t)N * 4);
    int*      deg  = (int*)     carve((size_t)N * 4);
    float*    A    = (float*)   carve((size_t)NF * 4);   // p1; later p2+agg2
    float*    B    = (float*)   carve((size_t)NF * 4);   // agg1
    uint16_t* xhi  = (uint16_t*)carve((size_t)NF * 2);
    uint16_t* xlo  = (uint16_t*)carve((size_t)NF * 2);
    uint32_t* w1h  = (uint32_t*)carve(8192 * 4);
    uint32_t* w1l  = (uint32_t*)carve(8192 * 4);
    uint32_t* w2h  = (uint32_t*)carve(4096 * 4);
    uint32_t* w2l  = (uint32_t*)carve(4096 * 4);
    float*    pool = (float*)   carve(64 * 64 * 4);
    float*    cnt  = (float*)   carve(64 * 4);
    float* p2   = A;                 // [N,64]
    float* agg2 = A + (size_t)NF2;   // [N,64]

    // normalization
    k_init<<<(N + 255) / 256, 256, 0, stream>>>(deg, pool, cnt, N);
    k_deg <<<(E + 255) / 256, 256, 0, stream>>>(edst, deg, E);
    k_dis <<<(N + 255) / 256, 256, 0, stream>>>(deg, dis, N);

    // inputs -> bf16 split; pack weights into WMMA B-fragment order
    k_split<<<(NF + 255) / 256, 256, 0, stream>>>(x, xhi, xlo, NF);
    k_pack <<<48, 256, 0, stream>>>(W1, W2, w1h, w1l, w2h, w2l);

    // layer 1
    k_gemm1<<<N / 16, 256, 0, stream>>>(xhi, xlo, w1h, w1l, dis, A, B, N);
    k_scatter128<<<(E * 32 + 255) / 256, 256, 0, stream>>>(esrc, edst, A, B, E);
    k_fin1 <<<(NF + 255) / 256, 256, 0, stream>>>(B, dis, b1, xhi, xlo, N);

    // layer 2
    k_gemm2<<<N / 32, 256, 0, stream>>>(xhi, xlo, w2h, w2l, dis, p2, agg2, N);
    k_scatter64<<<(E * 16 + 255) / 256, 256, 0, stream>>>(esrc, edst, p2, agg2, E);

    // pooling + head
    k_cnt  <<<(N + 255) / 256, 256, 0, stream>>>(batch, cnt, N);
    k_fin2 <<<(NF2 + 255) / 256, 256, 0, stream>>>(agg2, dis, b2, batch, pool, N);
    k_final<<<1, 64, 0, stream>>>(pool, cnt, lw, lb, out, G);
}
